// Model_35897336660253
// MI455X (gfx1250) — matrix operand order repaired and checked by reading, not compiled, not run on hardware
//
#include <hip/hip_runtime.h>
#include <stdint.h>
#include <stddef.h>

// ---------------- problem constants ----------------
constexpr int Bn  = 8;      // batch
constexpr int Np  = 8192;   // points
constexpr int Tq  = 2048;   // queries
constexpr int CWc = 1024;   // encoder width
constexpr int DWd = 256;    // decoder width
constexpr int ROWS = Bn * Tq;          // 16384 decoder rows
constexpr float EPSf = 1e-5f;

// ---------------- WMMA types ----------------
typedef __attribute__((ext_vector_type(16))) _Float16 v16h;
typedef __attribute__((ext_vector_type(8)))  float    v8f;
typedef __attribute__((ext_vector_type(4)))  unsigned u32x4;
// generic-vector int4: matches the async builtin's "V4i" parameter type exactly
typedef int i32x4g __attribute__((vector_size(16)));

union Frag16 { u32x4 u[2]; v16h h; };
union Acc8   { v8f v; float f[8]; };

// ---------------- CDNA5 async global->LDS path (guarded) ----------------
#define GAS1 __attribute__((address_space(1)))
#define GAS3 __attribute__((address_space(3)))
#if defined(__has_builtin)
#  if __has_builtin(__builtin_amdgcn_global_load_async_to_lds_b128) && \
      __has_builtin(__builtin_amdgcn_s_wait_asynccnt)
#    define USE_ASYNC_LDS 1
#  endif
#  if __has_builtin(__builtin_amdgcn_sched_barrier)
#    define SCHED_FENCE() __builtin_amdgcn_sched_barrier(0)
#  endif
#endif
#ifndef USE_ASYNC_LDS
#  define USE_ASYNC_LDS 0
#endif
#ifndef SCHED_FENCE
#  define SCHED_FENCE()
#endif

// =====================================================================
// k_init: zero the atomically-accumulated buffers (every launch)
// =====================================================================
__global__ void k_init(unsigned* c_bits, float* stats) {
    int i   = blockIdx.x * blockDim.x + threadIdx.x;
    int tot = gridDim.x * blockDim.x;
    for (int j = i; j < Bn * CWc;  j += tot) c_bits[j] = 0u;
    for (int j = i; j < 11 * 512;  j += tot) stats[j]  = 0.f;
}

// =====================================================================
// k_p99: per-batch 99th percentile (kth-value) of point norms.
// Radix-select on float bits (non-negative floats order like uints).
// =====================================================================
__global__ __launch_bounds__(256) void k_p99(const float* __restrict__ pts,
                                             float* p99, float* invp) {
    __shared__ unsigned keys[Np];   // 32 KB
    __shared__ int scnt[256];
    const int b = blockIdx.x, tid = threadIdx.x;

    int nv_local = 0;
    for (int i = tid; i < Np; i += 256) {
        const float* p = pts + ((size_t)b * Np + i) * 3;
        float x = p[0], y = p[1], z = p[2];
        bool valid = x > -50.f;
        float nr = sqrtf(x * x + y * y + z * z);
        keys[i] = valid ? __float_as_uint(nr) : 0x7F800000u;   // +inf if masked
        nv_local += valid ? 1 : 0;
    }
    scnt[tid] = nv_local;
    __syncthreads();
    for (int s = 128; s > 0; s >>= 1) { if (tid < s) scnt[tid] += scnt[tid + s]; __syncthreads(); }
    const int nvalid = scnt[0];
    __syncthreads();

    // k = 1 + round(0.99*(n_valid-1)), RNE rounding like jnp.round
    const int k = 1 + (int)rintf(0.99f * (float)(nvalid - 1));

    unsigned prefix = 0;
    int krem = k;
    for (int bit = 31; bit >= 0; --bit) {
        const unsigned pmask = (bit == 31) ? 0u : (0xFFFFFFFFu << (bit + 1));
        int cnt = 0;
        for (int i = tid; i < Np; i += 256) {
            unsigned v = keys[i];
            if (((v ^ prefix) & pmask) == 0u && ((v >> bit) & 1u) == 0u) cnt++;
        }
        scnt[tid] = cnt;
        __syncthreads();
        for (int s = 128; s > 0; s >>= 1) { if (tid < s) scnt[tid] += scnt[tid + s]; __syncthreads(); }
        const int cnt0 = scnt[0];
        if (krem > cnt0) { krem -= cnt0; prefix |= (1u << bit); }
        __syncthreads();
    }
    if (tid == 0) {
        float v = __uint_as_float(prefix);
        p99[b]  = v;
        invp[b] = 1.f / v;
    }
}

// =====================================================================
// k_transpose_f16: src[R][C] fp32 -> dst[C][R] f16
// =====================================================================
__global__ void k_transpose_f16(const float* __restrict__ src, _Float16* __restrict__ dst,
                                int R, int C) {
    int idx = blockIdx.x * blockDim.x + threadIdx.x;
    int tot = gridDim.x * blockDim.x;
    int n = R * C;
    for (; idx < n; idx += tot) {
        int r = idx / C, c = idx % C;
        dst[(size_t)c * R + r] = (_Float16)src[idx];
    }
}

// =====================================================================
// k_encoder: fused pointwise MLP (6->1024->1024) + masked max pool.
// Block = 64 rows x 256 cols; 8 waves = 4 mtiles x 2 nchunks(128 cols).
// Double-buffered LDS panels, one barrier per K-step, preloaded B frags.
// =====================================================================
__global__ __launch_bounds__(256)
void k_encoder(const float* __restrict__ pts, const float* __restrict__ rgb,
               const float* __restrict__ invp,
               const float* __restrict__ W1, const float* __restrict__ b1,
               const _Float16* __restrict__ W2t, const float* __restrict__ b2,
               unsigned* __restrict__ c_bits) {
    __shared__ __align__(16) _Float16 lA[2][64][32];     // 8 KB  h1 K-slice
    __shared__ __align__(16) _Float16 lBt[2][256][32];   // 32 KB W2t K-panels
    __shared__ float xf[64][6];
    __shared__ int   rvalid[64];

    const int tid     = threadIdx.x;
    const int bidx    = blockIdx.x >> 9;          // 512 blocks per batch
    const int rem     = blockIdx.x & 511;
    const int rb      = rem >> 2;                 // 128 row-blocks
    const int cq      = rem & 3;                  // col quarter
    const int row0    = rb * 64;
    const int colbase = cq * 256;

    if (tid < 64) {
        int r = row0 + tid;
        const float* pp = pts + ((size_t)bidx * Np + r) * 3;
        const float* rp = rgb + ((size_t)bidx * Np + r) * 3;
        float s = invp[bidx];
        float x = pp[0];
        rvalid[tid] = (x > -50.f) ? 1 : 0;
        xf[tid][0] = x * s; xf[tid][1] = pp[1] * s; xf[tid][2] = pp[2] * s;
        xf[tid][3] = rp[0]; xf[tid][4] = rp[1];    xf[tid][5] = rp[2];
    }
    __syncthreads();

    const int wv   = tid >> 5;
    const int lane = tid & 31;
    const int mtile = wv >> 1;                    // 0..3
    const int nc    = (wv & 1) * 128;             // local col chunk (8 N-tiles)
    const int kkme  = tid & 31;
    const int rbase = tid >> 5;

    // ---- stage lA[p] (h1 K-slice): one k per thread, 8 rows each
    auto stageA = [&](int p, int ks) {
        const int k = ks * 32 + kkme;
        float w0 = W1[0 * CWc + k], w1 = W1[1 * CWc + k], w2 = W1[2 * CWc + k];
        float w3 = W1[3 * CWc + k], w4 = W1[4 * CWc + k], w5 = W1[5 * CWc + k];
        float bb = b1[k];
        #pragma unroll
        for (int j = 0; j < 8; j++) {
            int r = rbase + j * 8;
            float a = bb + xf[r][0] * w0 + xf[r][1] * w1 + xf[r][2] * w2
                         + xf[r][3] * w3 + xf[r][4] * w4 + xf[r][5] * w5;
            lA[p][r][kkme] = (_Float16)fmaxf(a, 0.f);
        }
    };
    // ---- stage lBt[p] (W2t K-panel): one 32-half row per thread
    auto stageB = [&](int p, int ks) {
        const _Float16* src = W2t + (size_t)(colbase + tid) * CWc + ks * 32;
#if USE_ASYNC_LDS
        __builtin_amdgcn_global_load_async_to_lds_b128(
            (GAS1 i32x4g*)src, (GAS3 i32x4g*)&lBt[p][tid][0], 0, 0);
        __builtin_amdgcn_global_load_async_to_lds_b128(
            (GAS1 i32x4g*)src, (GAS3 i32x4g*)&lBt[p][tid][0], 16, 0);
        __builtin_amdgcn_global_load_async_to_lds_b128(
            (GAS1 i32x4g*)src, (GAS3 i32x4g*)&lBt[p][tid][0], 32, 0);
        __builtin_amdgcn_global_load_async_to_lds_b128(
            (GAS1 i32x4g*)src, (GAS3 i32x4g*)&lBt[p][tid][0], 48, 0);
#else
        const u32x4* s = (const u32x4*)src;
        u32x4* d = (u32x4*)&lBt[p][tid][0];
        d[0] = s[0]; d[1] = s[1]; d[2] = s[2]; d[3] = s[3];
        if (ks + 1 < 32) __builtin_prefetch((const void*)(s + 4), 0, 1);
#endif
    };

    v8f acc[8];
    #pragma unroll
    for (int i = 0; i < 8; i++) acc[i] = v8f{};

    stageA(0, 0);
    stageB(0, 0);

    for (int ks = 0; ks < 32; ks++) {
        const int p = ks & 1;
#if USE_ASYNC_LDS
        __builtin_amdgcn_s_wait_asynccnt(0);     // my panel[p] copies are done
#endif
        __syncthreads();                          // everyone's panel[p] + lA[p] visible

        // A fragment (16x32 f16): lane<16 -> K{0..7,16..23}; else K{8..15,24..31}
        Frag16 af;
        {
            int M   = mtile * 16 + (lane & 15);
            int ka0 = (lane < 16) ? 0 : 8;
            af.u[0] = *(const u32x4*)&lA[p][M][ka0];
            af.u[1] = *(const u32x4*)&lA[p][M][ka0 + 16];
        }
        // preload all 8 B fragments, then dense WMMA burst
        Frag16 bfr[8];
        const int kb0 = (lane < 16) ? 0 : 8;
        #pragma unroll
        for (int nt = 0; nt < 8; nt++) {
            int col = nc + nt * 16 + (lane & 15);
            bfr[nt].u[0] = *(const u32x4*)&lBt[p][col][kb0];
            bfr[nt].u[1] = *(const u32x4*)&lBt[p][col][kb0 + 16];
        }
        // keep all frag loads issued before the first WMMA (distinct registers)
        SCHED_FENCE();
        #pragma unroll
        for (int nt = 0; nt < 8; nt++) {
            acc[nt] = __builtin_amdgcn_wmma_f32_16x16x32_f16(
                false, af.h, false, bfr[nt].h, (short)0, acc[nt], false, false);
        }

        // stage next K-step into the other buffer (no barrier needed here)
        if (ks + 1 < 32) {
            stageA(p ^ 1, ks + 1);
            stageB(p ^ 1, ks + 1);
        }
    }

    // ---- epilogue: bias + relu + masked max over rows -> atomic max into c
    #pragma unroll
    for (int nt = 0; nt < 8; nt++) {
        int col_g = colbase + nc + nt * 16 + (lane & 15);
        float bb = b2[col_g];
        Acc8 a; a.v = acc[nt];
        float m = -1.f;                     // relu output >= 0, so -1 marks "none valid"
        #pragma unroll
        for (int v = 0; v < 8; v++) {
            int M = mtile * 16 + ((lane < 16) ? v : 8 + v);
            if (rvalid[M]) m = fmaxf(m, fmaxf(a.f[v] + bb, 0.f));
        }
        float o = __shfl_xor(m, 16, 32);    // merge row halves (same column)
        m = fmaxf(m, o);
        if (lane < 16 && m >= 0.f)
            atomicMax(&c_bits[(size_t)bidx * CWc + col_g], __float_as_uint(m));
    }
}

// =====================================================================
// k_matvec8: out[b][col] = c[b]·W[:,col] + bias[col]   (8x1024 @ 1024x256)
// =====================================================================
__global__ __launch_bounds__(256)
void k_matvec8(const float* __restrict__ c, const float* __restrict__ W,
               const float* __restrict__ bias, float* __restrict__ out) {
    const int b = blockIdx.x, col = threadIdx.x;
    const float* cb = c + (size_t)b * CWc;
    float acc = bias[col];
    for (int k = 0; k < CWc; k++) acc += cb[k] * W[(size_t)k * DWd + col];
    out[(size_t)b * DWd + col] = acc;
}

// =====================================================================
// k_fcp: net = (qc * inv_p99) @ fc_p_W + fc_p_b
// =====================================================================
__global__ void k_fcp(const float* __restrict__ qc, const float* __restrict__ invp,
                      const float* __restrict__ W, const float* __restrict__ bias,
                      float* __restrict__ net) {
    int idx = blockIdx.x * 256 + threadIdx.x;      // ROWS*DWd threads
    int r = idx >> 8, col = idx & 255;
    int b = r >> 11;                               // / Tq
    float s = invp[b];
    const float* q = qc + (size_t)r * 3;
    float acc = bias[col] + q[0] * s * W[0 * DWd + col]
                          + q[1] * s * W[1 * DWd + col]
                          + q[2] * s * W[2 * DWd + col];
    net[idx] = acc;
}

// =====================================================================
// k_stats: per-feature sum / sumsq over all 16384 rows (atomic accumulate)
// =====================================================================
__global__ __launch_bounds__(256)
void k_stats(const float* __restrict__ X, float* __restrict__ sums) {
    const int f  = threadIdx.x;
    const int r0 = blockIdx.x * 256;
    float s = 0.f, s2 = 0.f;
    for (int r = r0; r < r0 + 256; r++) {
        float v = X[(size_t)r * DWd + f];
        s += v; s2 += v * v;
    }
    atomicAdd(&sums[f], s);
    atomicAdd(&sums[256 + f], s2);
}

// =====================================================================
// k_cbn_mm: out = relu(CBN(X)) @ W + bias  (+ residual)
// Block = 32 rows x 256 cols; 8 waves = 2 mtiles x 4 nchunks(64 cols).
// Double-buffered LDS, preloaded B fragments.
// =====================================================================
__global__ __launch_bounds__(256)
void k_cbn_mm(const float* __restrict__ X, const float* __restrict__ sums,
              const float* __restrict__ gamma, const float* __restrict__ beta,
              const _Float16* __restrict__ Wt, const float* __restrict__ bias,
              const float* __restrict__ residual, float* __restrict__ out) {
    __shared__ __align__(16) _Float16 lA[2][32][32];     // 4 KB
    __shared__ __align__(16) _Float16 lBt[2][256][32];   // 32 KB
    __shared__ float meanv[DWd], invs[DWd], gv[DWd], bv[DWd];

    const int tid   = threadIdx.x;
    const int row0  = blockIdx.x * 32;
    const int batch = row0 >> 11;                    // blocks never straddle a batch

    {   // per-feature CBN params
        const int f = tid;
        float mean = sums[f] * (1.f / (float)ROWS);
        float ex2  = sums[256 + f] * (1.f / (float)ROWS);
        float var  = ex2 - mean * mean;
        meanv[f] = mean;
        invs[f]  = rsqrtf(var + EPSf);
        gv[f] = gamma[(size_t)batch * DWd + f];
        bv[f] = beta [(size_t)batch * DWd + f];
    }
    __syncthreads();

    const int wv    = tid >> 5;
    const int lane  = tid & 31;
    const int mtile = wv >> 2;            // 0..1
    const int nc    = (wv & 3) * 64;      // 4 N-tiles per wave
    const int kkme  = tid & 31;
    const int rbase = tid >> 5;

    auto stageA = [&](int p, int ks) {    // CBN-normalize into f16 LDS slice
        const int f = ks * 32 + kkme;
        float mg = gv[f], mb = bv[f], mm = meanv[f], mi = invs[f];
        #pragma unroll
        for (int j = 0; j < 4; j++) {
            int r = rbase + j * 8;
            float vx = X[(size_t)(row0 + r) * DWd + f];
            float aa = mg * ((vx - mm) * mi) + mb;
            lA[p][r][kkme] = (_Float16)fmaxf(aa, 0.f);
        }
    };
    auto stageB = [&](int p, int ks) {    // Wt K-panel (row stride 256 halves)
        const _Float16* src = Wt + (size_t)tid * DWd + ks * 32;
#if USE_ASYNC_LDS
        __builtin_amdgcn_global_load_async_to_lds_b128(
            (GAS1 i32x4g*)src, (GAS3 i32x4g*)&lBt[p][tid][0], 0, 0);
        __builtin_amdgcn_global_load_async_to_lds_b128(
            (GAS1 i32x4g*)src, (GAS3 i32x4g*)&lBt[p][tid][0], 16, 0);
        __builtin_amdgcn_global_load_async_to_lds_b128(
            (GAS1 i32x4g*)src, (GAS3 i32x4g*)&lBt[p][tid][0], 32, 0);
        __builtin_amdgcn_global_load_async_to_lds_b128(
            (GAS1 i32x4g*)src, (GAS3 i32x4g*)&lBt[p][tid][0], 48, 0);
#else
        const u32x4* s = (const u32x4*)src;
        u32x4* d = (u32x4*)&lBt[p][tid][0];
        d[0] = s[0]; d[1] = s[1]; d[2] = s[2]; d[3] = s[3];
#endif
    };

    v8f acc[4];
    #pragma unroll
    for (int i = 0; i < 4; i++) acc[i] = v8f{};

    stageA(0, 0);
    stageB(0, 0);

    for (int ks = 0; ks < 8; ks++) {
        const int p = ks & 1;
#if USE_ASYNC_LDS
        __builtin_amdgcn_s_wait_asynccnt(0);
#endif
        __syncthreads();

        Frag16 af;
        {
            int M   = mtile * 16 + (lane & 15);
            int ka0 = (lane < 16) ? 0 : 8;
            af.u[0] = *(const u32x4*)&lA[p][M][ka0];
            af.u[1] = *(const u32x4*)&lA[p][M][ka0 + 16];
        }
        Frag16 bfr[4];
        const int kb0 = (lane < 16) ? 0 : 8;
        #pragma unroll
        for (int nt = 0; nt < 4; nt++) {
            int col = nc + nt * 16 + (lane & 15);
            bfr[nt].u[0] = *(const u32x4*)&lBt[p][col][kb0];
            bfr[nt].u[1] = *(const u32x4*)&lBt[p][col][kb0 + 16];
        }
        SCHED_FENCE();
        #pragma unroll
        for (int nt = 0; nt < 4; nt++) {
            acc[nt] = __builtin_amdgcn_wmma_f32_16x16x32_f16(
                false, af.h, false, bfr[nt].h, (short)0, acc[nt], false, false);
        }

        if (ks + 1 < 8) {
            stageA(p ^ 1, ks + 1);
            stageB(p ^ 1, ks + 1);
        }
    }

    #pragma unroll
    for (int nt = 0; nt < 4; nt++) {
        int col = nc + nt * 16 + (lane & 15);
        float bb = bias[col];
        Acc8 a; a.v = acc[nt];
        #pragma unroll
        for (int v = 0; v < 8; v++) {
            int M = mtile * 16 + ((lane < 16) ? v : 8 + v);
            size_t o = (size_t)(row0 + M) * DWd + col;
            float val = a.f[v] + bb;
            if (residual) val += residual[o];
            out[o] = val;
        }
    }
}

// =====================================================================
// k_head: out[r] = relu(CBN(net[r])) · fc_out_W + fc_out_b
// =====================================================================
__global__ __launch_bounds__(256)
void k_head(const float* __restrict__ X, const float* __restrict__ sums,
            const float* __restrict__ gamma, const float* __restrict__ beta,
            const float* __restrict__ fcW, const float* __restrict__ fcB,
            float* __restrict__ out) {
    int r = blockIdx.x * 256 + threadIdx.x;
    int b = r >> 11;
    float acc = 0.f;
    for (int f = 0; f < DWd; f++) {
        float mean = sums[f] * (1.f / (float)ROWS);
        float var  = sums[256 + f] * (1.f / (float)ROWS) - mean * mean;
        float xn   = (X[(size_t)r * DWd + f] - mean) * rsqrtf(var + EPSf);
        float a    = gamma[(size_t)b * DWd + f] * xn + beta[(size_t)b * DWd + f];
        acc += fmaxf(a, 0.f) * fcW[f];
    }
    out[r] = acc + fcB[0];
}

// =====================================================================
// host: orchestration
// =====================================================================
extern "C" void kernel_launch(void* const* d_in, const int* in_sizes, int n_in,
                              void* d_out, int out_size, void* d_ws, size_t ws_size,
                              hipStream_t stream) {
    (void)in_sizes; (void)n_in; (void)out_size; (void)ws_size;

    const float* pts     = (const float*)d_in[0];
    const float* rgb     = (const float*)d_in[1];
    const float* qc      = (const float*)d_in[2];
    const float* enc_W1  = (const float*)d_in[3];
    const float* enc_b1  = (const float*)d_in[4];
    const float* enc_W2  = (const float*)d_in[5];
    const float* enc_b2  = (const float*)d_in[6];
    const float* fc_p_W  = (const float*)d_in[7];
    const float* fc_p_b  = (const float*)d_in[8];
    const float* blk_Wg0 = (const float*)d_in[9];
    const float* blk_bg0 = (const float*)d_in[10];
    const float* blk_Wb0 = (const float*)d_in[11];
    const float* blk_bb0 = (const float*)d_in[12];
    const float* blk_W0  = (const float*)d_in[13];
    const float* blk_b0  = (const float*)d_in[14];
    const float* blk_Wg1 = (const float*)d_in[15];
    const float* blk_bg1 = (const float*)d_in[16];
    const float* blk_Wb1 = (const float*)d_in[17];
    const float* blk_bb1 = (const float*)d_in[18];
    const float* blk_W1  = (const float*)d_in[19];
    const float* blk_b1  = (const float*)d_in[20];
    const float* bn_Wg   = (const float*)d_in[21];
    const float* bn_bg   = (const float*)d_in[22];
    const float* bn_Wb   = (const float*)d_in[23];
    const float* bn_bb   = (const float*)d_in[24];
    const float* fc_out_W= (const float*)d_in[25];
    const float* fc_out_b= (const float*)d_in[26];

    // ---- workspace carve (256B aligned) ----
    char* w = (char*)d_ws;
    auto carve = [&](size_t bytes) -> char* {
        char* p = w; w += (bytes + 255) & ~(size_t)255; return p;
    };
    float*    invp   = (float*)   carve(Bn * 4);
    float*    p99    = (float*)   carve(Bn * 4);
    unsigned* c_bits = (unsigned*)carve((size_t)Bn * CWc * 4);
    _Float16* W2t    = (_Float16*)carve((size_t)CWc * CWc * 2);
    _Float16* W0t    = (_Float16*)carve((size_t)5 * DWd * DWd * 2);
    _Float16* W1t    = (_Float16*)carve((size_t)5 * DWd * DWd * 2);
    float*    gammaA = (float*)   carve((size_t)11 * Bn * DWd * 4);
    float*    betaA  = (float*)   carve((size_t)11 * Bn * DWd * 4);
    float*    stats  = (float*)   carve((size_t)11 * 512 * 4);
    float*    net    = (float*)   carve((size_t)ROWS * DWd * 4);
    float*    dx     = (float*)   carve((size_t)ROWS * DWd * 4);

    // ---- preprocessing ----
    k_init<<<64, 256, 0, stream>>>(c_bits, stats);
    k_p99<<<Bn, 256, 0, stream>>>(pts, p99, invp);
    k_transpose_f16<<<2048, 256, 0, stream>>>(enc_W2, W2t, CWc, CWc);
    for (int i = 0; i < 5; i++) {
        k_transpose_f16<<<128, 256, 0, stream>>>(blk_W0 + (size_t)i * DWd * DWd,
                                                 W0t + (size_t)i * DWd * DWd, DWd, DWd);
        k_transpose_f16<<<128, 256, 0, stream>>>(blk_W1 + (size_t)i * DWd * DWd,
                                                 W1t + (size_t)i * DWd * DWd, DWd, DWd);
    }

    // ---- encoder (WMMA) ----
    k_encoder<<<Bn * 512, 256, 0, stream>>>(pts, rgb, invp, enc_W1, enc_b1,
                                            W2t, enc_b2, c_bits);
    const float* cfeat = (const float*)c_bits;   // max of relu >= 0: bits are the floats

    // ---- conditional gamma/beta (11 CBN layers) ----
    for (int i = 0; i < 5; i++) {
        k_matvec8<<<Bn, 256, 0, stream>>>(cfeat, blk_Wg0 + (size_t)i * CWc * DWd,
                                          blk_bg0 + (size_t)i * DWd,
                                          gammaA + (size_t)(2 * i) * Bn * DWd);
        k_matvec8<<<Bn, 256, 0, stream>>>(cfeat, blk_Wb0 + (size_t)i * CWc * DWd,
                                          blk_bb0 + (size_t)i * DWd,
                                          betaA + (size_t)(2 * i) * Bn * DWd);
        k_matvec8<<<Bn, 256, 0, stream>>>(cfeat, blk_Wg1 + (size_t)i * CWc * DWd,
                                          blk_bg1 + (size_t)i * DWd,
                                          gammaA + (size_t)(2 * i + 1) * Bn * DWd);
        k_matvec8<<<Bn, 256, 0, stream>>>(cfeat, blk_Wb1 + (size_t)i * CWc * DWd,
                                          blk_bb1 + (size_t)i * DWd,
                                          betaA + (size_t)(2 * i + 1) * Bn * DWd);
    }
    k_matvec8<<<Bn, 256, 0, stream>>>(cfeat, bn_Wg, bn_bg, gammaA + (size_t)10 * Bn * DWd);
    k_matvec8<<<Bn, 256, 0, stream>>>(cfeat, bn_Wb, bn_bb, betaA  + (size_t)10 * Bn * DWd);

    // ---- decoder ----
    k_fcp<<<ROWS, 256, 0, stream>>>(qc, invp, fc_p_W, fc_p_b, net);
    for (int i = 0; i < 5; i++) {
        int L0 = 2 * i, L1 = 2 * i + 1;
        k_stats<<<ROWS / 256, 256, 0, stream>>>(net, stats + (size_t)L0 * 512);
        k_cbn_mm<<<ROWS / 32, 256, 0, stream>>>(net, stats + (size_t)L0 * 512,
                                                gammaA + (size_t)L0 * Bn * DWd,
                                                betaA  + (size_t)L0 * Bn * DWd,
                                                W0t + (size_t)i * DWd * DWd,
                                                blk_b0 + (size_t)i * DWd,
                                                nullptr, dx);
        k_stats<<<ROWS / 256, 256, 0, stream>>>(dx, stats + (size_t)L1 * 512);
        k_cbn_mm<<<ROWS / 32, 256, 0, stream>>>(dx, stats + (size_t)L1 * 512,
                                                gammaA + (size_t)L1 * Bn * DWd,
                                                betaA  + (size_t)L1 * Bn * DWd,
                                                W1t + (size_t)i * DWd * DWd,
                                                blk_b1 + (size_t)i * DWd,
                                                net, net);
    }
    k_stats<<<ROWS / 256, 256, 0, stream>>>(net, stats + (size_t)10 * 512);
    k_head<<<ROWS / 256, 256, 0, stream>>>(net, stats + (size_t)10 * 512,
                                           gammaA + (size_t)10 * Bn * DWd,
                                           betaA  + (size_t)10 * Bn * DWd,
                                           fc_out_W, fc_out_b, (float*)d_out);
}
